// MultiHeadAttention_52965536694583
// MI455X (gfx1250) — compile-verified, matrix operand order checked by reading, not run on hardware
//
#include <hip/hip_runtime.h>

// ---------------------------------------------------------------------------
// CDNA5 (gfx1250) fused multi-head attention, bf16 WMMA pipeline.
//   x,W (f32) --cast--> bf16 --WMMA GEMM--> Q,K,V [B,H,S,64] bf16
//   flash-attention (WMMA scores + online softmax + TDM V-tiles + WMMA PV)
//   attn (bf16) --WMMA GEMM--> d_out (f32)
// ---------------------------------------------------------------------------

typedef __attribute__((ext_vector_type(16))) __bf16        v16bf;
typedef __attribute__((ext_vector_type(8)))  float         v8f;
typedef __attribute__((ext_vector_type(4)))  unsigned int  u32x4;
typedef __attribute__((ext_vector_type(8)))  int           i32x8;
typedef __attribute__((ext_vector_type(4)))  int           i32x4;

union FragU { v16bf v; u32x4 q[2]; __bf16 e[16]; };

// Load a 16-element bf16 fragment as two 16-byte chunks (matches the
// 16-bit A/B VGPR layout: K = koff+e (e<8) and K = 16+koff+(e-8)).
__device__ __forceinline__ v16bf load_frag(const __bf16* p0, const __bf16* p1) {
  FragU u;
  u.q[0] = *reinterpret_cast<const u32x4*>(p0);
  u.q[1] = *reinterpret_cast<const u32x4*>(p1);
  return u.v;
}

__device__ __forceinline__ v8f wmma_bf16(v16bf a, v16bf b, v8f c) {
  // (neg_a, A, neg_b, B, c_mod, C, reuse_a, reuse_b)
  return __builtin_amdgcn_wmma_f32_16x16x32_bf16(false, a, false, b,
                                                 (short)0, c, false, false);
}

// ---------------------------------------------------------------------------
// TDM: DMA a 2D bf16 tile (rows x 64, row stride 64 elems) from global memory
// into LDS at byte offset lds_off.  Descriptor per CDNA5 ISA
// section 8 (D# group0/group1); groups 2/3 zero (2D tensor).
// All descriptor dwords forced into SGPRs via readfirstlane.
// This toolchain exposes the 6-arg builtin form:
//   (u32x4 g0, i32x8 g1, i32x4 g2, i32x4 g3, i32x8, i32 cpol)
// ---------------------------------------------------------------------------
__device__ __forceinline__ void tdm_load_rows_bf16(unsigned lds_off,
                                                   const __bf16* gptr,
                                                   int rows) {
  const unsigned long long ga = (unsigned long long)(size_t)gptr;
  const unsigned ga_lo = (unsigned)(ga & 0xffffffffu);
  const unsigned ga_hi = (unsigned)((ga >> 32) & 0x1ffffffu);

  u32x4 g0;
  g0[0] = (unsigned)__builtin_amdgcn_readfirstlane(1);              // count=1
  g0[1] = (unsigned)__builtin_amdgcn_readfirstlane((int)lds_off);   // lds_addr
  g0[2] = (unsigned)__builtin_amdgcn_readfirstlane((int)ga_lo);     // gaddr lo
  g0[3] = (unsigned)__builtin_amdgcn_readfirstlane(
              (int)(ga_hi | (2u << 30)));                           // type=2

  i32x8 g1;
  g1[0] = __builtin_amdgcn_readfirstlane(1 << 16);       // data_size=2B
  g1[1] = __builtin_amdgcn_readfirstlane(64 << 16);      // tensor_dim0 = 64
  g1[2] = __builtin_amdgcn_readfirstlane(rows << 16);    // tensor_dim1 = rows
  g1[3] = __builtin_amdgcn_readfirstlane(64 << 16);      // tile_dim0 = 64
  g1[4] = __builtin_amdgcn_readfirstlane(rows);          // tile_dim1 = rows
  g1[5] = __builtin_amdgcn_readfirstlane(64);            // dim0_stride = 64
  g1[6] = 0;
  g1[7] = 0;

  i32x4 gz4 = {0, 0, 0, 0};
  i32x8 gz8 = {0, 0, 0, 0, 0, 0, 0, 0};
  __builtin_amdgcn_tensor_load_to_lds(g0, g1, gz4, gz4, gz8, 0);
}

// ---------------------------------------------------------------------------
// f32 -> bf16 cast
// ---------------------------------------------------------------------------
__global__ void cast_f32_to_bf16(const float* __restrict__ in,
                                 __bf16* __restrict__ out, int n) {
  int i = blockIdx.x * blockDim.x + threadIdx.x;
  int stride = gridDim.x * blockDim.x;
  for (; i < n; i += stride) out[i] = (__bf16)in[i];
}

// ---------------------------------------------------------------------------
// GEMM: Y[m,n] = sum_k A[m,k] * Bm[n,k]   (i.e. A @ Bm^T, both row-major)
// mode 0: f32 output row-major [M,N]
// mode 1: bf16 output, head-split layout [B,H,S,64] (S,H given; B = M/S)
// Block: 256 threads = 8 waves (2 M x 4 N), wave tile 64x64, WG tile 128x256.
// ---------------------------------------------------------------------------
__global__ __launch_bounds__(256)
void gemm_bf16_wmma(const __bf16* __restrict__ A, const __bf16* __restrict__ Bm,
                    float* __restrict__ Of32, __bf16* __restrict__ Obf,
                    int M, int N, int K, int mode, int S, int H) {
  const int tid  = threadIdx.x;
  const int wave = tid >> 5;
  const int lane = tid & 31;
  const int hi   = lane >> 4;       // which 16-lane half
  const int l16  = lane & 15;
  const int koff = hi * 8;          // per-half K offset in 16-bit fragments
  const int mwave = blockIdx.y * 128 + (wave >> 2) * 64;
  const int nwave = blockIdx.x * 256 + (wave & 3) * 64;

  v8f acc[4][4] = {};

  for (int kk = 0; kk < K; kk += 32) {
    v16bf af[4], bf[4];
#pragma unroll
    for (int i = 0; i < 4; ++i) {
      const __bf16* pa = A + (size_t)(mwave + i * 16 + l16) * K + kk + koff;
      af[i] = load_frag(pa, pa + 16);
      __builtin_prefetch(pa + 64, 0, 1);   // global_prefetch_b8, next k-tiles
    }
#pragma unroll
    for (int j = 0; j < 4; ++j) {
      const __bf16* pb = Bm + (size_t)(nwave + j * 16 + l16) * K + kk + koff;
      bf[j] = load_frag(pb, pb + 16);
      __builtin_prefetch(pb + 64, 0, 1);
    }
#pragma unroll
    for (int i = 0; i < 4; ++i)
#pragma unroll
      for (int j = 0; j < 4; ++j)
        acc[i][j] = wmma_bf16(af[i], bf[j], acc[i][j]);
  }

  if (mode == 0) {
#pragma unroll
    for (int i = 0; i < 4; ++i)
#pragma unroll
      for (int j = 0; j < 4; ++j) {
        const int n = nwave + j * 16 + l16;
#pragma unroll
        for (int r = 0; r < 8; ++r) {
          const int m = mwave + i * 16 + r + hi * 8;   // C/D row layout
          Of32[(size_t)m * N + n] = acc[i][j][r];
        }
      }
  } else {
#pragma unroll
    for (int i = 0; i < 4; ++i)
#pragma unroll
      for (int j = 0; j < 4; ++j) {
        const int n = nwave + j * 16 + l16;
        const int h = n >> 6;
        const int d = n & 63;
#pragma unroll
        for (int r = 0; r < 8; ++r) {
          const int m = mwave + i * 16 + r + hi * 8;
          const int b = m / S;
          const int s = m - b * S;
          Obf[(((size_t)b * H + h) * S + s) * 64 + d] = (__bf16)acc[i][j][r];
        }
      }
  }
}

// ---------------------------------------------------------------------------
// Flash attention for one (b,h): streaming softmax over 32-key chunks.
// Block: 128 threads = 4 waves; each wave owns 16 query rows (WG: 64 rows).
// V tiles are DMA'd into LDS by the Tensor Data Mover (TENSORcnt path).
// Q/K/V are bf16 [B*H, S, 64]; output merged bf16 [B, S, 1024].
// ---------------------------------------------------------------------------
__global__ __launch_bounds__(128)
void flash_attn_wmma(const __bf16* __restrict__ Q, const __bf16* __restrict__ Kx,
                     const __bf16* __restrict__ V, __bf16* __restrict__ Oa,
                     int S) {
  __shared__ __bf16 Vr[32][64];        // V chunk row-major (TDM destination)
  __shared__ __bf16 Vt[64][32];        // V chunk transposed: [d][key]
  __shared__ __bf16 Pt[4][16][32];     // per-wave P tile in A-layout [q][key]

  const int tid  = threadIdx.x;
  const int wave = __builtin_amdgcn_readfirstlane(tid >> 5);
  const int lane = tid & 31;
  const int hi   = lane >> 4;
  const int l16  = lane & 15;
  const int koff = hi * 8;
  const int bh   = blockIdx.y;                   // b*16 + h
  const int qbase = blockIdx.x * 64 + wave * 16;
  const float SCALE = 0.125f;                    // 1/sqrt(64)

  const __bf16* Qh = Q  + (size_t)bh * S * 64;
  const __bf16* Kh = Kx + (size_t)bh * S * 64;
  const __bf16* Vh = V  + (size_t)bh * S * 64;

  // LDS byte offset of Vr (low 32 bits of the generic LDS address)
  const unsigned vr_off = (unsigned)(size_t)&Vr[0][0];

  // Q A-fragments (row = l16, contraction d in [0,32) and [32,64))
  const __bf16* pq = Qh + (size_t)(qbase + l16) * 64;
  const v16bf qf0 = load_frag(pq + koff,      pq + koff + 16);
  const v16bf qf1 = load_frag(pq + 32 + koff, pq + 32 + koff + 16);

  v8f o[4] = {};
  float mrun[8], lrun[8];
#pragma unroll
  for (int r = 0; r < 8; ++r) { mrun[r] = -3.0e38f; lrun[r] = 0.0f; }

  for (int kb = 0; kb < S; kb += 32) {
    __syncthreads();   // previous iteration's Vr/Vt/Pt reads complete

    // Each wave DMAs its 8-key quarter of the V chunk into Vr via the TDM.
    tdm_load_rows_bf16(vr_off + (unsigned)(wave * 8) * 64u * 2u,
                       Vh + (size_t)(kb + wave * 8) * 64, 8);
    __builtin_amdgcn_s_wait_tensorcnt(0);
    __syncthreads();   // all quarters of Vr landed

    { // LDS->LDS transpose Vr -> Vt so B-fragments stay contiguous b128 reads
      const int key = tid >> 2;             // 0..31
      const int dq  = (tid & 3) * 16;       // 0,16,32,48
      FragU u;
      u.q[0] = *reinterpret_cast<const u32x4*>(&Vr[key][dq]);
      u.q[1] = *reinterpret_cast<const u32x4*>(&Vr[key][dq + 8]);
#pragma unroll
      for (int e = 0; e < 16; ++e) Vt[dq + e][key] = u.e[e];
    }

    // scores: S[q, kb+j*16+n] = Q . K^T  (B lane n = key, contraction = d)
    v8f s[2] = {};
#pragma unroll
    for (int j = 0; j < 2; ++j) {
      const __bf16* pk = Kh + (size_t)(kb + j * 16 + l16) * 64;
      v16bf kf0 = load_frag(pk + koff,      pk + koff + 16);
      v16bf kf1 = load_frag(pk + 32 + koff, pk + 32 + koff + 16);
      s[j] = wmma_bf16(qf0, kf0, s[j]);
      s[j] = wmma_bf16(qf1, kf1, s[j]);
    }

    // online softmax; row M = r + 8*hi lives across 16 lanes -> shfl_xor
#pragma unroll
    for (int r = 0; r < 8; ++r) {
      float a = s[0][r] * SCALE;
      float b = s[1][r] * SCALE;
      float mx = fmaxf(a, b);
#pragma unroll
      for (int off = 8; off >= 1; off >>= 1)
        mx = fmaxf(mx, __shfl_xor(mx, off, 32));
      const float mnew = fmaxf(mrun[r], mx);
      const float al   = __expf(mrun[r] - mnew);
      const float p0   = __expf(a - mnew);
      const float p1   = __expf(b - mnew);
      float rs = p0 + p1;
#pragma unroll
      for (int off = 8; off >= 1; off >>= 1)
        rs += __shfl_xor(rs, off, 32);
      lrun[r] = lrun[r] * al + rs;
      mrun[r] = mnew;
#pragma unroll
      for (int t = 0; t < 4; ++t) o[t][r] = o[t][r] * al;
      const int row = r + hi * 8;
      Pt[wave][row][l16]      = (__bf16)p0;   // C-layout -> A-layout via LDS
      Pt[wave][row][16 + l16] = (__bf16)p1;
    }

    __syncthreads();   // Vt + Pt visible

    // P A-fragment (row = l16, contraction = key in chunk)
    const __bf16* pp = &Pt[wave][l16][0];
    const v16bf pf = load_frag(pp + koff, pp + koff + 16);
#pragma unroll
    for (int t = 0; t < 4; ++t) {
      // V B-fragment: lane n = d = t*16+l16, contraction = key (Vt row)
      const __bf16* pv = &Vt[t * 16 + l16][0];
      v16bf vf = load_frag(pv + koff, pv + koff + 16);
      o[t] = wmma_bf16(pf, vf, o[t]);
    }
  }

  // normalize and store merged-head bf16 output [B, S, 1024]
  const int b = bh >> 4;
  const int h = bh & 15;
#pragma unroll
  for (int r = 0; r < 8; ++r) {
    const float inv = 1.0f / lrun[r];
    const int qrow = qbase + r + hi * 8;
    const size_t base = ((size_t)(b * S + qrow)) * 1024 + (size_t)h * 64;
#pragma unroll
    for (int t = 0; t < 4; ++t)
      Oa[base + t * 16 + l16] = (__bf16)(o[t][r] * inv);
  }
}

// ---------------------------------------------------------------------------
// Host launcher
// ---------------------------------------------------------------------------
extern "C" void kernel_launch(void* const* d_in, const int* in_sizes, int n_in,
                              void* d_out, int out_size, void* d_ws, size_t ws_size,
                              hipStream_t stream) {
  const float* x  = (const float*)d_in[0];
  const float* Wq = (const float*)d_in[1];
  const float* Wk = (const float*)d_in[2];
  const float* Wv = (const float*)d_in[3];
  const float* Wo = (const float*)d_in[4];

  const int B = 2, S = 2048, D = 1024, H = 16;
  const int M = B * S;                 // 4096 rows
  const size_t MD = (size_t)M * D;     // 4M elems
  const size_t DD = (size_t)D * D;     // 1M elems

  // workspace carve-up (all bf16; total 24M elems = 48 MB)
  __bf16* Xb  = (__bf16*)d_ws;
  __bf16* Wqb = Xb  + MD;
  __bf16* Wkb = Wqb + DD;
  __bf16* Wvb = Wkb + DD;
  __bf16* Wob = Wvb + DD;
  __bf16* Qh  = Wob + DD;   // [B,H,S,64]
  __bf16* Kh  = Qh  + MD;
  __bf16* Vh  = Kh  + MD;
  __bf16* Ab  = Vh  + MD;   // merged attn output [B,S,1024]

  cast_f32_to_bf16<<<dim3(2048), dim3(256), 0, stream>>>(x,  Xb,  (int)MD);
  cast_f32_to_bf16<<<dim3(512),  dim3(256), 0, stream>>>(Wq, Wqb, (int)DD);
  cast_f32_to_bf16<<<dim3(512),  dim3(256), 0, stream>>>(Wk, Wkb, (int)DD);
  cast_f32_to_bf16<<<dim3(512),  dim3(256), 0, stream>>>(Wv, Wvb, (int)DD);
  cast_f32_to_bf16<<<dim3(512),  dim3(256), 0, stream>>>(Wo, Wob, (int)DD);

  const dim3 gblk(256);
  const dim3 ggrid(D / 256, M / 128);  // (4, 32)

  // Q/K/V projections -> head-split bf16
  gemm_bf16_wmma<<<ggrid, gblk, 0, stream>>>(Xb, Wqb, nullptr, Qh, M, D, D, 1, S, H);
  gemm_bf16_wmma<<<ggrid, gblk, 0, stream>>>(Xb, Wkb, nullptr, Kh, M, D, D, 1, S, H);
  gemm_bf16_wmma<<<ggrid, gblk, 0, stream>>>(Xb, Wvb, nullptr, Vh, M, D, D, 1, S, H);

  // streaming attention (TDM-fed V tiles)
  flash_attn_wmma<<<dim3(S / 64, B * H), dim3(128), 0, stream>>>(Qh, Kh, Vh, Ab, S);

  // output projection -> f32 d_out
  gemm_bf16_wmma<<<ggrid, gblk, 0, stream>>>(Ab, Wob, (float*)d_out, nullptr,
                                             M, D, D, 0, S, H);
}
